// Resonator_927712935965
// MI455X (gfx1250) — 8-bit matrix operand order repaired and checked by reading, compiled, not run on hardware
//
#include <hip/hip_runtime.h>
#include <stdint.h>


namespace {

constexpr int kB = 128;
constexpr int kF = 4;
constexpr int kV = 2048;
constexpr int kD = 4096;
constexpr int kIters = 20;

typedef int v8i __attribute__((ext_vector_type(8)));
typedef int v4i __attribute__((ext_vector_type(4)));
typedef unsigned int u32;
typedef unsigned char u8;
typedef signed char i8;

__device__ __forceinline__ void async_tile_b128(u32 lds_addr, const void* gaddr) {
  asm volatile("global_load_async_to_lds_b128 %0, %1, off"
               :: "v"(lds_addr), "v"(gaddr)
               : "memory");
}
__device__ __forceinline__ void wait_async0() {
  asm volatile("s_wait_asynccnt 0x0" ::: "memory");
}

__global__ void init_flags_kernel(int* flags) {
  int i = threadIdx.x;
  if (i < kIters) flags[i] = 0;
}

__global__ void pack_inputs_kernel(const float* __restrict__ inp, u8* __restrict__ im) {
  int i = blockIdx.x * blockDim.x + threadIdx.x;
  if (i < kB * kD) im[i] = (inp[i] < 0.0f) ? 0xFFu : 0x00u;
}

__global__ void pack_est_kernel(const float* __restrict__ est, u8* __restrict__ estm) {
  int i = blockIdx.x * blockDim.x + threadIdx.x;
  if (i >= kB * kF * kD) return;
  int b = i / (kF * kD);
  int rem = i - b * (kF * kD);
  int f = rem / kD;
  int d = rem - f * kD;
  estm[((size_t)f * kB + b) * kD + d] = (est[i] < 0.0f) ? 0xFFu : 0x00u;
}

__global__ void pack_cb_kernel(const float* __restrict__ cbf, i8* __restrict__ cb,
                               i8* __restrict__ cbT) {
  int i = blockIdx.x * blockDim.x + threadIdx.x;
  if (i >= kF * kV * kD) return;
  int f = i / (kV * kD);
  int rem = i - f * (kV * kD);
  int v = rem / kD;
  int d = rem - v * kD;
  i8 s = (cbf[i] >= 0.0f) ? (i8)1 : (i8)(-1);
  cb[i] = s;
  cbT[((size_t)f * kD + d) * kV + v] = s;
}

__global__ void base_kernel(const u32* __restrict__ im, const u32* __restrict__ estm,
                            u32* __restrict__ base) {
  int i = blockIdx.x * blockDim.x + threadIdx.x;
  constexpr int N = kB * kD / 4;
  if (i >= N) return;
  base[i] = im[i] ^ estm[i] ^ estm[i + N] ^ estm[i + 2 * N] ^ estm[i + 3 * N];
}

template <bool kFinal>
__global__ __launch_bounds__(256) void gemm1_kernel(
    const u32* __restrict__ base, const u8* __restrict__ estm,
    const i8* __restrict__ cb,
    int* __restrict__ sim, u8* __restrict__ simlo, i8* __restrict__ simhi) {
  __shared__ u8 smem[2 * 4096];

  const int tid = threadIdx.x;
  const int lane = tid & 31;
  const int wave = tid >> 5;
  const int f = blockIdx.x >> 5;
  const int n0 = (blockIdx.x & 31) << 6;
  const int m0 = wave << 4;
  const int half = lane >> 4;
  const int lrow = lane & 15;
  const int aRow = m0 + lrow;

  const int scol = tid >> 2;
  const int schunk = (tid & 3) << 4;
  const i8* gsrc = cb + ((size_t)f * kV + n0 + scol) * kD + schunk;
  const u32 ldsStage = (u32)(size_t)(&smem[0]) + (u32)(scol * 64 + schunk);

  async_tile_b128(ldsStage, gsrc);
  wait_async0();
  __syncthreads();

  const u32* baseRow = base + (size_t)aRow * (kD / 4);
  const u32* estRow = (const u32*)(estm + ((size_t)f * kB + aRow) * kD);

  v8i acc[4];
#pragma unroll
  for (int t = 0; t < 4; ++t) acc[t] = (v8i){0, 0, 0, 0, 0, 0, 0, 0};

  auto compute = [&](int k0, int bsel) {
    const int kb = (k0 >> 2) + 2 * half;
    v8i A;
#pragma unroll
    for (int j = 0; j < 4; ++j) {
      u32 x0 = estRow[kb + 4 * j];
      u32 x1 = estRow[kb + 4 * j + 1];
      if (!kFinal) {
        x0 ^= baseRow[kb + 4 * j];
        x1 ^= baseRow[kb + 4 * j + 1];
      }
      A[2 * j] = (int)(x0 | 0x01010101u);
      A[2 * j + 1] = (int)(x1 | 0x01010101u);
    }
    const u8* bb = &smem[bsel * 4096];
#pragma unroll
    for (int t = 0; t < 4; ++t) {
      const int* bcol = (const int*)(bb + (16 * t + lrow) * 64 + 8 * half);
      v4i p = (v4i){bcol[0], bcol[1], bcol[4], bcol[5]};
      v4i q = (v4i){bcol[8], bcol[9], bcol[12], bcol[13]};
      v8i Bm;
      Bm[0] = p[0]; Bm[1] = p[1]; Bm[2] = p[2]; Bm[3] = p[3];
      Bm[4] = q[0]; Bm[5] = q[1]; Bm[6] = q[2]; Bm[7] = q[3];
      acc[t] = __builtin_amdgcn_wmma_i32_16x16x64_iu8(true, A, true, Bm, acc[t],
                                                      false, false);
    }
  };

  int buf = 0;
#pragma unroll 1
  for (int k0 = 0; k0 < kD - 64; k0 += 64) {
    async_tile_b128(ldsStage + (u32)((buf ^ 1) * 4096), gsrc + k0 + 64);
    compute(k0, buf);
    wait_async0();
    __syncthreads();
    buf ^= 1;
  }
  compute(kD - 64, buf);

#pragma unroll
  for (int t = 0; t < 4; ++t) {
#pragma unroll
    for (int rr = 0; rr < 8; ++rr) {
      const int brow_ = m0 + rr + 8 * half;
      const size_t off = ((size_t)f * kB + brow_) * kV + (n0 + 16 * t + lrow);
      int s = acc[t][rr];
      if (kFinal) {
        sim[off] = s;
      } else {
        simlo[off] = (u8)(s & 0xFF);
        simhi[off] = (i8)(s >> 8);
      }
    }
  }
}

__global__ __launch_bounds__(256) void gemm2_kernel(
    const u8* __restrict__ simlo, const i8* __restrict__ simhi,
    const i8* __restrict__ cbT,
    u8* __restrict__ estm, int* __restrict__ flag) {
  __shared__ u8 smem[2 * 4096];

  const int tid = threadIdx.x;
  const int lane = tid & 31;
  const int wave = tid >> 5;
  const int f = blockIdx.x >> 6;
  const int n0 = (blockIdx.x & 63) << 6;
  const int m0 = wave << 4;
  const int half = lane >> 4;
  const int lrow = lane & 15;
  const int aRow = m0 + lrow;

  const int scol = tid >> 2;
  const int schunk = (tid & 3) << 4;
  const i8* gsrc = cbT + ((size_t)f * kD + n0 + scol) * kV + schunk;
  const u32 ldsStage = (u32)(size_t)(&smem[0]) + (u32)(scol * 64 + schunk);

  async_tile_b128(ldsStage, gsrc);
  wait_async0();
  __syncthreads();

  const u32* lo32 = (const u32*)(simlo + ((size_t)f * kB + aRow) * kV);
  const u32* hi32 = (const u32*)(simhi + ((size_t)f * kB + aRow) * kV);

  v8i accL[4], accH[4];
#pragma unroll
  for (int t = 0; t < 4; ++t) {
    accL[t] = (v8i){0, 0, 0, 0, 0, 0, 0, 0};
    accH[t] = (v8i){0, 0, 0, 0, 0, 0, 0, 0};
  }

  auto compute = [&](int k0, int bsel) {
    const int kb = (k0 >> 2) + 2 * half;
    v8i AL, AH;
#pragma unroll
    for (int j = 0; j < 4; ++j) {
      AL[2 * j] = (int)lo32[kb + 4 * j];
      AL[2 * j + 1] = (int)lo32[kb + 4 * j + 1];
      AH[2 * j] = (int)hi32[kb + 4 * j];
      AH[2 * j + 1] = (int)hi32[kb + 4 * j + 1];
    }
    const u8* bb = &smem[bsel * 4096];
#pragma unroll
    for (int t = 0; t < 4; ++t) {
      const int* bcol = (const int*)(bb + (16 * t + lrow) * 64 + 8 * half);
      v4i p = (v4i){bcol[0], bcol[1], bcol[4], bcol[5]};
      v4i q = (v4i){bcol[8], bcol[9], bcol[12], bcol[13]};
      v8i Bm;
      Bm[0] = p[0]; Bm[1] = p[1]; Bm[2] = p[2]; Bm[3] = p[3];
      Bm[4] = q[0]; Bm[5] = q[1]; Bm[6] = q[2]; Bm[7] = q[3];
      accL[t] = __builtin_amdgcn_wmma_i32_16x16x64_iu8(false, AL, true, Bm,
                                                       accL[t], false, false);
      accH[t] = __builtin_amdgcn_wmma_i32_16x16x64_iu8(true, AH, true, Bm,
                                                       accH[t], false, false);
    }
  };

  int buf = 0;
#pragma unroll 1
  for (int k0 = 0; k0 < kV - 64; k0 += 64) {
    async_tile_b128(ldsStage + (u32)((buf ^ 1) * 4096), gsrc + k0 + 64);
    compute(k0, buf);
    wait_async0();
    __syncthreads();
    buf ^= 1;
  }
  compute(kV - 64, buf);

  int changed = 0;
#pragma unroll
  for (int t = 0; t < 4; ++t) {
#pragma unroll
    for (int rr = 0; rr < 8; ++rr) {
      const int brow_ = m0 + rr + 8 * half;
      const int dcol = n0 + 16 * t + lrow;
      int s = accH[t][rr] * 256 + accL[t][rr];
      u8 m = (s >= 0) ? 0x00u : 0xFFu;
      const size_t off = ((size_t)f * kB + brow_) * kD + dcol;
      u8 old = estm[off];
      changed |= (int)(old != m);
      estm[off] = m;
    }
  }
  if (changed) atomicOr(flag, 1);
}

__global__ __launch_bounds__(256) void argmax_kernel(const int* __restrict__ sim,
                                                     float* __restrict__ out) {
  const int lane = threadIdx.x & 31;
  const int wid = blockIdx.x * 8 + (threadIdx.x >> 5);
  const int b = wid >> 2;
  const int f = wid & 3;
  const int* row = sim + ((size_t)f * kB + b) * kV;
  int bestAbs = -1, bestIdx = 0;
  for (int v = lane; v < kV; v += 32) {
    int s = row[v];
    int a = (s < 0) ? -s : s;
    if (a > bestAbs) { bestAbs = a; bestIdx = v; }
  }
  for (int offm = 16; offm >= 1; offm >>= 1) {
    int oa = __shfl_xor(bestAbs, offm, 32);
    int oi = __shfl_xor(bestIdx, offm, 32);
    if (oa > bestAbs || (oa == bestAbs && oi < bestIdx)) {
      bestAbs = oa; bestIdx = oi;
    }
  }
  if (lane == 0) out[b * kF + f] = (float)bestIdx;
}

__global__ void k_kernel(const int* __restrict__ flags, float* __restrict__ out) {
  if (threadIdx.x == 0 && blockIdx.x == 0) {
    int k = 0;
    bool done = false;
    for (int i = 0; i < kIters; ++i) {
      if (!done) k = i;
      done = done || (flags[i] == 0);
    }
    out[kB * kF] = (float)k;
  }
}

}

extern "C" void kernel_launch(void* const* d_in, const int* in_sizes, int n_in,
                              void* d_out, int out_size, void* d_ws, size_t ws_size,
                              hipStream_t stream) {
  (void)in_sizes; (void)n_in; (void)out_size; (void)ws_size;
  const float* inputs = (const float*)d_in[0];
  const float* init_est = (const float*)d_in[1];
  const float* codebooks = (const float*)d_in[2];
  float* out = (float*)d_out;

  char* ws = (char*)d_ws;
  size_t off = 0;
  auto alloc = [&](size_t bytes) -> char* {
    char* p = ws + off;
    off += (bytes + 255) & ~(size_t)255;
    return p;
  };
  i8* cb = (i8*)alloc((size_t)kF * kV * kD);
  i8* cbT = (i8*)alloc((size_t)kF * kV * kD);
  u8* estm = (u8*)alloc((size_t)kF * kB * kD);
  u8* im = (u8*)alloc((size_t)kB * kD);
  u32* base = (u32*)alloc((size_t)kB * kD);
  int* sim = (int*)alloc((size_t)kF * kB * kV * 4);
  u8* simlo = (u8*)alloc((size_t)kF * kB * kV);
  i8* simhi = (i8*)alloc((size_t)kF * kB * kV);
  int* flags = (int*)alloc(kIters * sizeof(int));

  init_flags_kernel<<<1, 32, 0, stream>>>(flags);
  pack_inputs_kernel<<<(kB * kD + 255) / 256, 256, 0, stream>>>(inputs, im);
  pack_est_kernel<<<(kB * kF * kD + 255) / 256, 256, 0, stream>>>(init_est, estm);
  pack_cb_kernel<<<(kF * kV * kD + 255) / 256, 256, 0, stream>>>(codebooks, cb, cbT);

  for (int it = 0; it < kIters; ++it) {
    base_kernel<<<(kB * kD / 4 + 255) / 256, 256, 0, stream>>>(
        (const u32*)im, (const u32*)estm, base);
    gemm1_kernel<false><<<128, 256, 0, stream>>>(base, estm, cb, sim, simlo, simhi);
    gemm2_kernel<<<256, 256, 0, stream>>>(simlo, simhi, cbT, estm, flags + it);
  }
  gemm1_kernel<true><<<128, 256, 0, stream>>>(base, estm, cb, sim, simlo, simhi);
  argmax_kernel<<<64, 256, 0, stream>>>(sim, out);
  k_kernel<<<1, 1, 0, stream>>>(flags, out);
}
